// SharedAttention_27023934227125
// MI455X (gfx1250) — matrix operand order repaired and checked by reading, not compiled, not run on hardware
//
#include <hip/hip_runtime.h>
#include <stdint.h>

typedef __bf16 bf16;
typedef __attribute__((ext_vector_type(16))) bf16  v16bf;
typedef __attribute__((ext_vector_type(8)))  bf16  v8bf;
typedef __attribute__((ext_vector_type(4)))  bf16  v4bf;
typedef __attribute__((ext_vector_type(8)))  float v8f;
typedef __attribute__((ext_vector_type(4))) unsigned int u32x4;
typedef __attribute__((ext_vector_type(4))) int  i32x4;
typedef __attribute__((ext_vector_type(8))) int  i32x8;

#define B_    2
#define T_    2048
#define C_    1024
#define H_    16
#define D_    64
#define RANK_ 8
#define SCALE_ 2.0f            // alpha/rank = 16/8
#define NROWS (B_ * T_)        // 4096

#if __has_builtin(__builtin_amdgcn_tensor_load_to_lds)
#define HAVE_TDM 1
#else
#define HAVE_TDM 0
#endif

static __device__ __forceinline__ v8f vzero8() {
  v8f v;
#pragma unroll
  for (int i = 0; i < 8; ++i) v[i] = 0.0f;
  return v;
}

// A fragment 16(M)x32(K) bf16 from row-major [M,K] tile (ISA 7.12.2 layout):
// lanes 0-15: M=lane, K = {0..7, 16..23}; lanes 16-31: M=lane-16, K = {8..15, 24..31}
static __device__ __forceinline__ v16bf load_fragA(const bf16* base, int ld) {
  const int lane = threadIdx.x & 31;
  const int m    = lane & 15;
  const int ks   = (lane < 16) ? 0 : 8;
  const bf16* p  = base + (size_t)m * ld + ks;
  v8bf lo = *(const v8bf*)(p);
  v8bf hi = *(const v8bf*)(p + 16);
  v16bf r;
#pragma unroll
  for (int i = 0; i < 8; ++i) { r[i] = lo[i]; r[i + 8] = hi[i]; }
  return r;
}

// B fragment 32(K)x16(N) bf16 sourced from row-major [N,K] tile:
// lane holds N=lane%16, K = lane<16 ? {0..7, 16..23} : {8..15, 24..31}
static __device__ __forceinline__ v16bf load_fragB(const bf16* base, int ld) {
  const int lane = threadIdx.x & 31;
  const int n    = lane & 15;
  const int kb   = (lane < 16) ? 0 : 8;
  const bf16* p  = base + (size_t)n * ld + kb;
  v8bf lo = *(const v8bf*)(p);
  v8bf hi = *(const v8bf*)(p + 16);
  v16bf r;
#pragma unroll
  for (int i = 0; i < 8; ++i) { r[i] = lo[i]; r[i + 8] = hi[i]; }
  return r;
}

// B fragment read from an LDS tile filled by the Tensor Data Mover. The TDM's
// LDS write is invisible to LLVM's memory model (it folded plain/volatile
// generic loads of the "never-stored" shared array), so issue native
// ds_load_b128s via inline asm on the 32-bit LDS offset (generic shared ptr =
// src_shared_base | as3_offset; low 32 bits are the DS address per ISA 10.2).
static __device__ __forceinline__ v16bf load_fragB_lds(const bf16* base, int ld) {
  const int lane = threadIdx.x & 31;
  const int n    = lane & 15;
  const int kb   = (lane < 16) ? 0 : 8;
  const unsigned off = (unsigned)(uintptr_t)(base + (size_t)n * ld + kb);
  v8bf lo, hi;
  __asm__ __volatile__(
      "ds_load_b128 %0, %2\n\t"
      "ds_load_b128 %1, %2 offset:32\n\t"
      "s_wait_dscnt 0x0"
      : "=&v"(lo), "=&v"(hi)
      : "v"(off)
      : "memory");
  v16bf r;
#pragma unroll
  for (int i = 0; i < 8; ++i) { r[i] = lo[i]; r[i + 8] = hi[i]; }
  return r;
}

#define WMMA_BF16(a, b, c) \
  __builtin_amdgcn_wmma_f32_16x16x32_bf16(false, (a), false, (b), (short)0, (c), false, false)

#if HAVE_TDM
// TDM: DMA a [128 rows x 32 elems] bf16 tile (row stride = K elems) into LDS.
// Descriptor per CDNA5 ISA ch.8: group0 = {flags, lds_addr, global_addr, type=2},
// group1 = {data_size=2B, tensor_dim0=K, tensor_dim1=N, tile_dim0=32, tile_dim1=128,
//           tensor_dim0_stride=K}. Groups 2/3 zero (2-D tensor).
static __device__ __forceinline__ void tdm_load_tile(const bf16* gsrc, unsigned lds_off,
                                                     int K, int N) {
  const unsigned long long ga = (unsigned long long)(uintptr_t)gsrc;
  u32x4 g0;
  g0[0] = 1u;                                  // count=1, user mode, no gather
  g0[1] = lds_off;                             // LDS byte address
  g0[2] = (unsigned)ga;                        // global_addr[31:0]
  g0[3] = (unsigned)(ga >> 32) | (2u << 30);   // global_addr[56:32] | type=2
  i32x8 g1;
  g1[0] = (int)(1u << 16);                                            // data_size=2B
  g1[1] = (int)(((unsigned)K & 0xffffu) << 16);                       // tensor_dim0 lo
  g1[2] = (int)((((unsigned)K >> 16) & 0xffffu) |
                (((unsigned)N & 0xffffu) << 16));                     // dim0 hi | dim1 lo
  g1[3] = (int)((((unsigned)N >> 16) & 0xffffu) | (32u << 16));       // dim1 hi | tile_dim0=32
  g1[4] = 128;                                                        // tile_dim1=128
  g1[5] = K;                                                          // tensor_dim0_stride lo32
  g1[6] = 0;
  g1[7] = 0;
  i32x4 z4 = {0, 0, 0, 0};
#if defined(__clang_major__) && (__clang_major__ >= 23)
  i32x8 z8 = {0, 0, 0, 0, 0, 0, 0, 0};
  __builtin_amdgcn_tensor_load_to_lds(g0, g1, z4, z4, z8, 0);
#else
  __builtin_amdgcn_tensor_load_to_lds(g0, g1, z4, z4, 0);
#endif
}
#endif

// ---------------- fp32 -> bf16 conversion ----------------
__global__ __launch_bounds__(256) void k_cvt(const float* __restrict__ s,
                                             bf16* __restrict__ d, int n) {
  int i = (blockIdx.x * 256 + threadIdx.x) * 4;
  if (i + 3 < n) {
    const float4 f = *(const float4*)(s + i);
    v4bf o = {(bf16)f.x, (bf16)f.y, (bf16)f.z, (bf16)f.w};
    *(v4bf*)(d + i) = o;
  }
}

// ---------------- LoRA low-rank projection: low[n,j] = sum_c src[n,c]*A[r,j,c] ----
__global__ __launch_bounds__(256) void k_low(const float* __restrict__ x32,
                                             const bf16* __restrict__ x16,
                                             const float* __restrict__ Aall,
                                             const int* __restrict__ ridx,
                                             float* __restrict__ low, int K) {
  const int lane = threadIdx.x & 31;
  const int g = blockIdx.x * 8 + (threadIdx.x >> 5);  // one wave per output
  const int n = g >> 3, j = g & 7;
  const int r = *ridx;
  const float* ar = Aall + ((size_t)r * RANK_ + j) * K;
  float s = 0.f;
  if (x32) {
    const float* xr = x32 + (size_t)n * K;
    for (int c = lane; c < K; c += 32) s += xr[c] * ar[c];
  } else {
    const bf16* xr = x16 + (size_t)n * K;
    for (int c = lane; c < K; c += 32) s += (float)xr[c] * ar[c];
  }
#pragma unroll
  for (int d = 16; d >= 1; d >>= 1) s += __shfl_xor(s, d, 32);
  if (lane == 0) low[(size_t)n * RANK_ + j] = s;
}

// ---------------- fused GEMM + bias + LoRA epilogue ----------------
// out = A[M,K] @ W[N,K]^T + bias[N] + SCALE * low[M,8] @ loraB[r][N,8]^T
// block = 256 thr = 8 waves (4M x 2N); wave tile 32x64; block tile 128x128
// W block tile (128x32 bf16, 8KB) staged into LDS by TDM, double buffered.
// mode 0: scatter bf16 to [B,H,T,D]; mode 1: scatter bf16 to [B,H,D,T]; mode 2: fp32 [M,N]
__global__ __launch_bounds__(256) void k_gemm_lora(
    const bf16* __restrict__ A, const bf16* __restrict__ W,
    const float* __restrict__ bias, const float* __restrict__ low,
    const float* __restrict__ loraB_all, const int* __restrict__ ridx,
    int K, int N, int mode, void* __restrict__ outp) {
  __shared__ __align__(128) bf16 wtile[2][128 * 32];
  const int lane  = threadIdx.x & 31;
  const int wave  = threadIdx.x >> 5;
  const int row0  = blockIdx.x * 128 + (wave & 3) * 32;
  const int bcol0 = blockIdx.y * 128;            // block's N origin
  const int wcol  = (wave >> 2) * 64;            // wave's N offset inside block tile

  v8f acc[2][4];
#pragma unroll
  for (int i = 0; i < 2; ++i)
#pragma unroll
    for (int j = 0; j < 4; ++j) acc[i][j] = vzero8();

  const int KT = K / 32;
#if HAVE_TDM
  if (wave == 0) tdm_load_tile(W + (size_t)bcol0 * K, 0u, K, N);
#endif

  for (int it = 0; it < KT; ++it) {
    const int k0 = it * 32;
    const int p  = it & 1;
#if HAVE_TDM
    if (wave == 0) {
      if (it + 1 < KT) {
        tdm_load_tile(W + (size_t)bcol0 * K + (k0 + 32),
                      (unsigned)(((it + 1) & 1) * (128 * 32 * 2)), K, N);
        __builtin_amdgcn_s_wait_tensorcnt(1);  // tile for THIS step is complete
      } else {
        __builtin_amdgcn_s_wait_tensorcnt(0);
      }
    }
#else
    // fallback: cooperative copy of the 8KB tile
    for (int c = threadIdx.x; c < 512; c += 256) {
      const int rr = c >> 2, co = (c & 3) * 8;
      *(v8bf*)&wtile[p][rr * 32 + co] = *(const v8bf*)(W + (size_t)(bcol0 + rr) * K + k0 + co);
    }
#endif
    __syncthreads();   // wtile[p] ready for all waves

    v16bf af[2], wf[4];
#pragma unroll
    for (int i = 0; i < 2; ++i)
      af[i] = load_fragA(A + (size_t)(row0 + 16 * i) * K + k0, K);
#pragma unroll
    for (int j = 0; j < 4; ++j)
      wf[j] = load_fragB_lds(&wtile[p][(wcol + j * 16) * 32], 32);
    if (k0 + 32 < K)
      __builtin_prefetch(A + (size_t)row0 * K + k0 + 32, 0, 3);
#pragma unroll
    for (int i = 0; i < 2; ++i)
#pragma unroll
      for (int j = 0; j < 4; ++j) acc[i][j] = WMMA_BF16(af[i], wf[j], acc[i][j]);

    __syncthreads();   // all waves done with wtile[p] before it is overwritten
  }

  const int r = *ridx;
  const float* lB = loraB_all + (size_t)r * N * RANK_;
  const int nl  = lane & 15;
  const int off = (lane < 16) ? 0 : 8;

  float bo[4], lb[4][RANK_];
#pragma unroll
  for (int j = 0; j < 4; ++j) {
    const int o = bcol0 + wcol + j * 16 + nl;
    bo[j] = bias[o];
#pragma unroll
    for (int q = 0; q < RANK_; ++q) lb[j][q] = lB[(size_t)o * RANK_ + q];
  }

#pragma unroll
  for (int i = 0; i < 2; ++i) {
#pragma unroll
    for (int e = 0; e < 8; ++e) {
      const int rrow = row0 + i * 16 + off + e;
      const float* lr = low + (size_t)rrow * RANK_;
      float lv[RANK_];
#pragma unroll
      for (int q = 0; q < RANK_; ++q) lv[q] = lr[q];
#pragma unroll
      for (int j = 0; j < 4; ++j) {
        const int o = bcol0 + wcol + j * 16 + nl;
        float s = 0.f;
#pragma unroll
        for (int q = 0; q < RANK_; ++q) s += lv[q] * lb[j][q];
        const float v = acc[i][j][e] + bo[j] + SCALE_ * s;
        if (mode == 2) {
          ((float*)outp)[(size_t)rrow * N + o] = v;
        } else {
          const int b = rrow >> 11, t = rrow & (T_ - 1);
          const int h = o >> 6, d = o & (D_ - 1);
          size_t addr;
          if (mode == 0) addr = (((size_t)(b * H_ + h)) * T_ + t) * D_ + d;
          else           addr = (((size_t)(b * H_ + h)) * D_ + d) * T_ + t;
          ((bf16*)outp)[addr] = (bf16)v;
        }
      }
    }
  }
}

// ---------------- flash attention: one wave per 16-query tile ----------------
__global__ __launch_bounds__(128) void k_attn(
    const bf16* __restrict__ qh, const bf16* __restrict__ kh,
    const bf16* __restrict__ vt, const float* __restrict__ mask,
    bf16* __restrict__ yh) {
  __shared__ __align__(16) bf16 pbuf[4][16 * 32];
  const int lane = threadIdx.x & 31;
  const int wave = threadIdx.x >> 5;
  const int g  = blockIdx.x * 4 + wave;     // 0 .. B*H*(T/16)-1
  const int bh = g >> 7;                    // 128 query tiles per (b,h)
  const int t0 = (g & 127) * 16;
  const int b = bh >> 4, h = bh & 15;

  const bf16* qb = qh + ((size_t)bh * T_ + t0) * D_;
  const bf16* kb = kh + (size_t)bh * T_ * D_;
  const bf16* vb = vt + (size_t)bh * D_ * T_;

  const v16bf aq0 = load_fragA(qb, D_);        // d = 0..31
  const v16bf aq1 = load_fragA(qb + 32, D_);   // d = 32..63

  const int nl  = lane & 15;
  const int off = (lane < 16) ? 0 : 8;

  float mrow[8], lrow[8];
#pragma unroll
  for (int e = 0; e < 8; ++e) { mrow[e] = -1e30f; lrow[e] = 0.f; }
  v8f yacc[4];
#pragma unroll
  for (int d = 0; d < 4; ++d) yacc[d] = vzero8();

  bf16* pb = pbuf[wave];

  for (int s0 = 0; s0 < t0 + 16; s0 += 32) {
    v8f sv[2];
#pragma unroll
    for (int hf = 0; hf < 2; ++hf) {
      sv[hf] = vzero8();
      const bf16* kr = kb + (size_t)(s0 + 16 * hf) * D_;
      sv[hf] = WMMA_BF16(aq0, load_fragB(kr, D_), sv[hf]);
      sv[hf] = WMMA_BF16(aq1, load_fragB(kr + 32, D_), sv[hf]);
    }
#pragma unroll
    for (int e = 0; e < 8; ++e) {
      const int t = t0 + off + e;
      const float m0 = mask[(size_t)t * T_ + s0 + nl];
      const float m1 = mask[(size_t)t * T_ + s0 + 16 + nl];
      const float a0 = sv[0][e] * 0.125f + m0;   // 1/sqrt(64)
      const float a1 = sv[1][e] * 0.125f + m1;
      float mx = fmaxf(a0, a1);
#pragma unroll
      for (int dl = 1; dl < 16; dl <<= 1) mx = fmaxf(mx, __shfl_xor(mx, dl, 32));
      const float nm = fmaxf(mrow[e], mx);
      const float fr = __expf(mrow[e] - nm);
      const float p0 = __expf(a0 - nm);
      const float p1 = __expf(a1 - nm);
      float ps = p0 + p1;
#pragma unroll
      for (int dl = 1; dl < 16; dl <<= 1) ps += __shfl_xor(ps, dl, 32);
      lrow[e] = lrow[e] * fr + ps;
      mrow[e] = nm;
#pragma unroll
      for (int d = 0; d < 4; ++d) yacc[d][e] *= fr;
      // stash P tile (C layout -> row-major bf16) for A-fragment reload
      pb[(off + e) * 32 + nl]      = (bf16)p0;
      pb[(off + e) * 32 + nl + 16] = (bf16)p1;
    }
    const v16bf ap = load_fragA(pb, 32);   // P as 16x32 A fragment via LDS
#pragma unroll
    for (int d = 0; d < 4; ++d)
      yacc[d] = WMMA_BF16(ap, load_fragB(vb + (size_t)(d * 16) * T_ + s0, T_), yacc[d]);
  }

#pragma unroll
  for (int e = 0; e < 8; ++e) {
    const float inv = 1.0f / lrow[e];
    const int t = t0 + off + e;
#pragma unroll
    for (int d = 0; d < 4; ++d)
      yh[((size_t)b * T_ + t) * C_ + h * D_ + d * 16 + nl] = (bf16)(yacc[d][e] * inv);
  }
}

// ---------------- host-side orchestration ----------------
extern "C" void kernel_launch(void* const* d_in, const int* in_sizes, int n_in,
                              void* d_out, int out_size, void* d_ws, size_t ws_size,
                              hipStream_t stream) {
  (void)in_sizes; (void)n_in; (void)out_size; (void)ws_size;
  const float* x    = (const float*)d_in[0];
  const int*   ridx = (const int*)d_in[1];
  const float* mask = (const float*)d_in[2];
  const float *Wt[4], *bt[4], *At[4], *Bt[4];
  for (int i = 0; i < 4; ++i) {           // q, k, v, o
    Wt[i] = (const float*)d_in[3 + 4 * i];
    bt[i] = (const float*)d_in[4 + 4 * i];
    At[i] = (const float*)d_in[5 + 4 * i];
    Bt[i] = (const float*)d_in[6 + 4 * i];
  }

  char* w = (char*)d_ws;
  bf16* xh = (bf16*)w;  w += (size_t)NROWS * C_ * 2;
  bf16* Wh[4];
  for (int i = 0; i < 4; ++i) { Wh[i] = (bf16*)w; w += (size_t)C_ * C_ * 2; }
  bf16* qh  = (bf16*)w; w += (size_t)NROWS * C_ * 2;   // [B,H,T,D]
  bf16* khb = (bf16*)w; w += (size_t)NROWS * C_ * 2;   // [B,H,T,D]
  bf16* vtb = (bf16*)w; w += (size_t)NROWS * C_ * 2;   // [B,H,D,T]
  bf16* yh  = (bf16*)w; w += (size_t)NROWS * C_ * 2;   // [B,T,C]
  float* lowv[4];
  for (int i = 0; i < 4; ++i) { lowv[i] = (float*)w; w += (size_t)NROWS * RANK_ * 4; }

  // 1) bf16 conversions
  k_cvt<<<(NROWS * C_) / 1024, 256, 0, stream>>>(x, xh, NROWS * C_);
  for (int i = 0; i < 4; ++i)
    k_cvt<<<(C_ * C_) / 1024, 256, 0, stream>>>(Wt[i], Wh[i], C_ * C_);

  // 2) LoRA low-rank projections of x (q,k,v)
  for (int i = 0; i < 3; ++i)
    k_low<<<NROWS, 256, 0, stream>>>(x, (const bf16*)nullptr, At[i], ridx, lowv[i], C_);

  // 3) fused QKV projections (TDM -> LDS -> WMMA)
  dim3 gg(NROWS / 128, C_ / 128);
  k_gemm_lora<<<gg, 256, 0, stream>>>(xh, Wh[0], bt[0], lowv[0], Bt[0], ridx, C_, C_, 0, qh);
  k_gemm_lora<<<gg, 256, 0, stream>>>(xh, Wh[1], bt[1], lowv[1], Bt[1], ridx, C_, C_, 0, khb);
  k_gemm_lora<<<gg, 256, 0, stream>>>(xh, Wh[2], bt[2], lowv[2], Bt[2], ridx, C_, C_, 1, vtb);

  // 4) flash attention (WMMA)
  k_attn<<<(B_ * H_ * (T_ / 16)) / 4, 128, 0, stream>>>(qh, khb, vtb, mask, yh);

  // 5) output projection with LoRA (WMMA) -> fp32 d_out
  k_low<<<NROWS, 256, 0, stream>>>((const float*)nullptr, yh, At[3], ridx, lowv[3], C_);
  k_gemm_lora<<<gg, 256, 0, stream>>>(yh, Wh[3], bt[3], lowv[3], Bt[3], ridx, C_, C_, 2, d_out);
}
